// LittleBitITQSpecLinear_8615704396413
// MI455X (gfx1250) — matrix operand order repaired and checked by reading, not compiled, not run on hardware
//
#include <hip/hip_runtime.h>

// ---------------------------------------------------------------------------
// Problem constants (from reference)
// ---------------------------------------------------------------------------
#define T_ROWS 8192   // B*S = 4*2048
#define DIN    4096
#define DOUT   4096
#define DSPLIT 1024

// Block tile
#define BM 128
#define BN 128
#define BK 32     // == WMMA bf16 K

typedef float  v8f   __attribute__((ext_vector_type(8)));
typedef __bf16 v16bf __attribute__((ext_vector_type(16)));
typedef int    v4i   __attribute__((vector_size(16)));   // matches builtin param

union Frag16 { v16bf v; uint4 u[2]; };   // 32 bytes
union Cvt16  { __bf16 h[16]; uint4 q[2]; };

#define AS1 __attribute__((address_space(1)))
#define AS3 __attribute__((address_space(3)))

// ---- CDNA5 async global->LDS copy (ASYNCcnt) with sync fallback -----------
#if defined(__has_builtin)
# if __has_builtin(__builtin_amdgcn_global_load_async_to_lds_b128)
#  define HAVE_ASYNC_LDS 1
# endif
#endif

#ifdef HAVE_ASYNC_LDS
// src: global (AS1), dst: LDS (AS3); AS-qualified ptrs implicitly convert to
// generic params (generic is a superset AS on AMDGPU), and match if qualified.
# define ASYNC_CP16(g, l) \
    __builtin_amdgcn_global_load_async_to_lds_b128((AS1 v4i*)(g), (AS3 v4i*)(l), 0, 0)
# if __has_builtin(__builtin_amdgcn_s_wait_asynccnt)
#  define ASYNC_WAIT0() __builtin_amdgcn_s_wait_asynccnt(0)
# else
#  define ASYNC_WAIT0() asm volatile("s_wait_asynccnt 0" ::: "memory")
# endif
#else
# define ASYNC_CP16(g, l) (*(uint4*)(l) = *(const uint4*)(g))
# define ASYNC_WAIT0()
#endif

static __device__ __forceinline__ float signf(float v) {
    return (v > 0.0f) ? 1.0f : ((v < 0.0f) ? -1.0f : 0.0f);
}

// ---------------------------------------------------------------------------
// Weight preprocessing (scales folded, transposed for contiguous B-frags):
//   Wv[i][s] = sign(V[s,i]) * v2[i]                 (bf16, [DIN  x DSPLIT])
//   Wu[s][o] = sign(U[o,s]) * v1[s]*u2[s]*u1[o]     (bf16, [DSPLIT x DOUT])
// ---------------------------------------------------------------------------
__global__ __launch_bounds__(256) void quant_v_kernel(
    const float* __restrict__ V, const float* __restrict__ v2,
    __bf16* __restrict__ Wv)
{
    size_t idx = (size_t)blockIdx.x * blockDim.x + threadIdx.x;  // over DIN*DSPLIT
    int i = (int)(idx / DSPLIT);
    int s = (int)(idx % DSPLIT);
    Wv[idx] = (__bf16)(signf(V[(size_t)s * DIN + i]) * v2[i]);
}

__global__ __launch_bounds__(256) void quant_u_kernel(
    const float* __restrict__ U, const float* __restrict__ v1,
    const float* __restrict__ u2, const float* __restrict__ u1,
    __bf16* __restrict__ Wu)
{
    size_t idx = (size_t)blockIdx.x * blockDim.x + threadIdx.x;  // over DSPLIT*DOUT
    int s = (int)(idx / DOUT);
    int o = (int)(idx % DOUT);
    Wu[idx] = (__bf16)(signf(U[(size_t)o * DSPLIT + s]) * v1[s] * u2[s] * u1[o]);
}

// ---------------------------------------------------------------------------
// GEMM1:  H (bf16 [T_ROWS x DSPLIT]) = X (f32, cvt) @ Wv
// 256 thr = 8 waves tiled 4(M) x 2(N); each wave: 2 M-frags x 4 N-frags
// = 8 WMMAs / K-step. A staged via VGPRs (needs f32->bf16 cvt), B staged
// via async-to-LDS. Double-buffered LDS, one barrier per K-step.
// ---------------------------------------------------------------------------
__global__ __launch_bounds__(256) void gemm1_kernel(
    const float*  __restrict__ X,
    const __bf16* __restrict__ W,   // [DIN x DSPLIT]
    __bf16*       __restrict__ H)   // [T_ROWS x DSPLIT]
{
    __shared__ __align__(16) __bf16 Alds[2][BM][BK];   // 2 x 8 KB
    __shared__ __align__(16) __bf16 Blds[2][BK][BN];   // 2 x 8 KB

    const int tid  = threadIdx.x;
    const int lane = tid & 31;
    const int wave = tid >> 5;
    const int wm   = wave >> 1;  // 0..3 : 32-row slice
    const int wn   = wave & 1;   // 0..1 : 64-col slice
    const int row0 = blockIdx.y * BM;
    const int col0 = blockIdx.x * BN;
    const int m_l  = lane & 15;
    const int hi   = lane >> 4;

    // staging coords
    const int arow = tid >> 1;         // 0..127
    const int acol = (tid & 1) * 16;   // 0 / 16
    const int brow = tid >> 3;         // 0..31
    const int bcol = (tid & 7) * 16;   // 0..112

    v8f acc[2][4];
#pragma unroll
    for (int mt = 0; mt < 2; ++mt)
#pragma unroll
        for (int nt = 0; nt < 4; ++nt) acc[mt][nt] = (v8f){0,0,0,0,0,0,0,0};

    const float* asrc = X + (size_t)(row0 + arow) * DIN + acol;
    float4 areg[4];

    auto loadA = [&](int k0) {
        const float4* s = (const float4*)(asrc + k0);
        areg[0] = s[0]; areg[1] = s[1]; areg[2] = s[2]; areg[3] = s[3];
    };
    auto issueB = [&](int k0, int buf) {
        const __bf16* g = W + (size_t)(k0 + brow) * DSPLIT + col0 + bcol;
        __bf16* l = &Blds[buf][brow][bcol];
        ASYNC_CP16(g, l);
        ASYNC_CP16(g + 8, l + 8);
    };

    loadA(0);
    issueB(0, 0);

    const int NIT = DIN / BK;   // 128
#pragma unroll 1
    for (int it = 0; it < NIT; ++it) {
        const int b = it & 1;
        // convert prefetched A regs -> bf16 -> LDS (current buffer)
        {
            Cvt16 c;
            const float* f = (const float*)areg;
#pragma unroll
            for (int j = 0; j < 16; ++j) c.h[j] = (__bf16)f[j];
            uint4* dst = (uint4*)&Alds[b][arow][acol];
            dst[0] = c.q[0];
            dst[1] = c.q[1];
        }
        ASYNC_WAIT0();
        __syncthreads();
        if (it + 1 < NIT) {          // overlap next tile's copy with compute
            issueB((it + 1) * BK, b ^ 1);
            loadA((it + 1) * BK);
        }
        // ---- fragments + 8 WMMAs ----
        Frag16 aF0, aF1;
        const __bf16* pa = &Alds[b][wm * 32 + m_l][0];
        aF0.u[0] = *(const uint4*)(pa + hi * 8);
        aF0.u[1] = *(const uint4*)(pa + 16 + hi * 8);
        const __bf16* pa1 = pa + 16 * BK;
        aF1.u[0] = *(const uint4*)(pa1 + hi * 8);
        aF1.u[1] = *(const uint4*)(pa1 + 16 + hi * 8);
#pragma unroll
        for (int nt = 0; nt < 4; ++nt) {
            Frag16 bF;
            const __bf16* pb = &Blds[b][0][wn * 64 + nt * 16 + m_l];
#pragma unroll
            for (int s = 0; s < 16; ++s)
                bF.v[s] = pb[((s & 8) * 2 + hi * 8 + (s & 7)) * BN];
            acc[0][nt] = __builtin_amdgcn_wmma_f32_16x16x32_bf16(
                false, aF0.v, false, bF.v, (short)0, acc[0][nt], false, false);
            acc[1][nt] = __builtin_amdgcn_wmma_f32_16x16x32_bf16(
                false, aF1.v, false, bF.v, (short)0, acc[1][nt], false, false);
        }
    }

    // epilogue: f32 acc -> bf16 H
#pragma unroll
    for (int mt = 0; mt < 2; ++mt)
#pragma unroll
        for (int nt = 0; nt < 4; ++nt) {
            const int col = col0 + wn * 64 + nt * 16 + m_l;     // n = lane&15
#pragma unroll
            for (int r = 0; r < 8; ++r) {
                const int row = row0 + wm * 32 + mt * 16 + r + hi * 8;
                H[(size_t)row * DSPLIT + col] = (__bf16)acc[mt][nt][r];
            }
        }
}

// ---------------------------------------------------------------------------
// GEMM2:  Y (f32) = H0 @ Wu0 + H1 @ Wu1 + bias   (u1 folded into Wu)
// Both branches flattened into one K loop (K_eff = 2048); fully async staging.
// ---------------------------------------------------------------------------
__global__ __launch_bounds__(256) void gemm2_kernel(
    const __bf16* __restrict__ H0, const __bf16* __restrict__ W0,
    const __bf16* __restrict__ H1, const __bf16* __restrict__ W1,
    const float*  __restrict__ bias,
    float*        __restrict__ Y)
{
    __shared__ __align__(16) __bf16 Alds[2][BM][BK];
    __shared__ __align__(16) __bf16 Blds[2][BK][BN];

    const int tid  = threadIdx.x;
    const int lane = tid & 31;
    const int wave = tid >> 5;
    const int wm   = wave >> 1;
    const int wn   = wave & 1;
    const int row0 = blockIdx.y * BM;
    const int col0 = blockIdx.x * BN;
    const int m_l  = lane & 15;
    const int hi   = lane >> 4;

    const int arow = tid >> 1;
    const int acol = (tid & 1) * 16;
    const int brow = tid >> 3;
    const int bcol = (tid & 7) * 16;

    v8f acc[2][4];
#pragma unroll
    for (int mt = 0; mt < 2; ++mt)
#pragma unroll
        for (int nt = 0; nt < 4; ++nt) acc[mt][nt] = (v8f){0,0,0,0,0,0,0,0};

    const int SEG = DSPLIT / BK;   // 32 tiles per branch

    auto issue = [&](int it, int buf) {
        const __bf16* Hh = (it >= SEG) ? H1 : H0;
        const __bf16* Wb = (it >= SEG) ? W1 : W0;
        const int k0 = (it & (SEG - 1)) * BK;
        {
            const __bf16* g = Hh + (size_t)(row0 + arow) * DSPLIT + k0 + acol;
            __bf16* l = &Alds[buf][arow][acol];
            ASYNC_CP16(g, l);
            ASYNC_CP16(g + 8, l + 8);
        }
        {
            const __bf16* g = Wb + (size_t)(k0 + brow) * DOUT + col0 + bcol;
            __bf16* l = &Blds[buf][brow][bcol];
            ASYNC_CP16(g, l);
            ASYNC_CP16(g + 8, l + 8);
        }
    };

    issue(0, 0);

    const int NIT = 2 * SEG;   // 64
#pragma unroll 1
    for (int it = 0; it < NIT; ++it) {
        const int b = it & 1;
        ASYNC_WAIT0();
        __syncthreads();
        if (it + 1 < NIT) issue(it + 1, b ^ 1);

        Frag16 aF0, aF1;
        const __bf16* pa = &Alds[b][wm * 32 + m_l][0];
        aF0.u[0] = *(const uint4*)(pa + hi * 8);
        aF0.u[1] = *(const uint4*)(pa + 16 + hi * 8);
        const __bf16* pa1 = pa + 16 * BK;
        aF1.u[0] = *(const uint4*)(pa1 + hi * 8);
        aF1.u[1] = *(const uint4*)(pa1 + 16 + hi * 8);
#pragma unroll
        for (int nt = 0; nt < 4; ++nt) {
            Frag16 bF;
            const __bf16* pb = &Blds[b][0][wn * 64 + nt * 16 + m_l];
#pragma unroll
            for (int s = 0; s < 16; ++s)
                bF.v[s] = pb[((s & 8) * 2 + hi * 8 + (s & 7)) * BN];
            acc[0][nt] = __builtin_amdgcn_wmma_f32_16x16x32_bf16(
                false, aF0.v, false, bF.v, (short)0, acc[0][nt], false, false);
            acc[1][nt] = __builtin_amdgcn_wmma_f32_16x16x32_bf16(
                false, aF1.v, false, bF.v, (short)0, acc[1][nt], false, false);
        }
    }

    // epilogue: + bias, f32 store
#pragma unroll
    for (int mt = 0; mt < 2; ++mt)
#pragma unroll
        for (int nt = 0; nt < 4; ++nt) {
            const int col = col0 + wn * 64 + nt * 16 + m_l;
            const float bval = bias[col];
#pragma unroll
            for (int r = 0; r < 8; ++r) {
                const int row = row0 + wm * 32 + mt * 16 + r + hi * 8;
                Y[(size_t)row * DOUT + col] = acc[mt][nt][r] + bval;
            }
        }
}

// ---------------------------------------------------------------------------
// Launch
// ---------------------------------------------------------------------------
extern "C" void kernel_launch(void* const* d_in, const int* in_sizes, int n_in,
                              void* d_out, int out_size, void* d_ws, size_t ws_size,
                              hipStream_t stream) {
    const float* x    = (const float*)d_in[0];
    const float* V    = (const float*)d_in[1];
    const float* U    = (const float*)d_in[2];
    const float* v2   = (const float*)d_in[3];
    const float* v1   = (const float*)d_in[4];
    const float* u2   = (const float*)d_in[5];
    const float* u1   = (const float*)d_in[6];
    const float* V_R  = (const float*)d_in[7];
    const float* U_R  = (const float*)d_in[8];
    const float* v2_R = (const float*)d_in[9];
    const float* v1_R = (const float*)d_in[10];
    const float* u2_R = (const float*)d_in[11];
    const float* u1_R = (const float*)d_in[12];
    const float* bias = (const float*)d_in[13];
    float* Y = (float*)d_out;

    // Workspace (bf16): Wv0,Wv1 (8 MB ea) | Wu0,Wu1 (8 MB ea) | H0,H1 (16 MB ea)
    __bf16* Wv0 = (__bf16*)d_ws;
    __bf16* Wv1 = Wv0 + (size_t)DIN * DSPLIT;
    __bf16* Wu0 = Wv1 + (size_t)DIN * DSPLIT;
    __bf16* Wu1 = Wu0 + (size_t)DSPLIT * DOUT;
    __bf16* H0  = Wu1 + (size_t)DSPLIT * DOUT;
    __bf16* H1  = H0  + (size_t)T_ROWS * DSPLIT;

    quant_v_kernel<<<(DIN * DSPLIT) / 256, 256, 0, stream>>>(V,   v2,   Wv0);
    quant_v_kernel<<<(DIN * DSPLIT) / 256, 256, 0, stream>>>(V_R, v2_R, Wv1);
    quant_u_kernel<<<(DSPLIT * DOUT) / 256, 256, 0, stream>>>(U,   v1,   u2,   u1,   Wu0);
    quant_u_kernel<<<(DSPLIT * DOUT) / 256, 256, 0, stream>>>(U_R, v1_R, u2_R, u1_R, Wu1);

    dim3 g1(DSPLIT / BN, T_ROWS / BM);   // (8, 64)
    gemm1_kernel<<<g1, 256, 0, stream>>>(x, Wv0, H0);
    gemm1_kernel<<<g1, 256, 0, stream>>>(x, Wv1, H1);

    dim3 g2(DOUT / BN, T_ROWS / BM);     // (32, 64)
    gemm2_kernel<<<g2, 256, 0, stream>>>(H0, Wu0, H1, Wu1, bias, Y);
}
